// RepresentationFunction_59811714564645
// MI455X (gfx1250) — compile-verified, matrix operand order checked by reading, not run on hardware
//
#include <hip/hip_runtime.h>
#include <math.h>

typedef __attribute__((ext_vector_type(16))) _Float16 v16h;
typedef __attribute__((ext_vector_type(8)))  float    v8f;
typedef __attribute__((ext_vector_type(2)))  _Float16 h2;

#define DMODEL 128
#define LHIST  200
#define LPAD   208          // 13 tiles of 16 rows
#define ROWW   130          // padded LDS row width (halves) -> 260B stride, conflict-free
#define NT_ROW 13
#define SCALE  0.08838834764831843f   // 1/sqrt(128)

// Branchless fast tanh: tanh(x) = 1 - 2/(exp(2x)+1)
//   x >> 0: e=inf -> rcp=0 -> +1 ;  x << 0: e=0 -> 1-2 = -1. No divergence.
__device__ __forceinline__ float fast_tanh(float x) {
    const float e = __expf(2.0f * x);                  // v_exp_f32 path
    return 1.0f - 2.0f * __builtin_amdgcn_rcpf(e + 1.0f);  // v_rcp_f32 + fma
}

__global__ __launch_bounds__(256) void din_attn_kernel(
    const int*   __restrict__ user_idx,
    const int*   __restrict__ item_idx,
    const int*   __restrict__ user_hist,
    const int*   __restrict__ item_hist,
    const float* __restrict__ user_embed_target,
    const float* __restrict__ user_embed_hist,
    const float* __restrict__ user_embed_global,
    const float* __restrict__ item_embed_target,
    const float* __restrict__ item_embed_hist,
    const float* __restrict__ item_embed_global,
    const float* __restrict__ proj_u_w, const float* __restrict__ proj_u_b,
    const float* __restrict__ proj_i_w, const float* __restrict__ proj_i_b,
    float* __restrict__ out, int Bn, int n_users, int n_items)
{
    __shared__ _Float16 sRef[LPAD * ROWW];   // gathered history values, f16
    __shared__ float    sScore[LPAD];
    __shared__ float    sMask[LPAD];
    __shared__ int      sIdx[LPAD];
    __shared__ float    sPartial[2 * DMODEL];
    __shared__ float    sInv[2];

    const int b    = blockIdx.x;
    const int side = blockIdx.y;          // 0 = user side, 1 = item side
    const int tid  = threadIdx.x;
    const int lane = tid & 31;
    const int wave = tid >> 5;

    int target, counterpart, pad_val;
    const int   *hist;
    const float *ehist, *q, *W, *bias, *etarget;
    if (side == 0) {
        target = user_idx[b]; counterpart = item_idx[b]; pad_val = n_items;
        hist = user_hist; ehist = item_embed_hist; q = user_embed_global;
        W = proj_u_w; bias = proj_u_b; etarget = user_embed_target;
    } else {
        target = item_idx[b]; counterpart = user_idx[b]; pad_val = n_users;
        hist = item_hist; ehist = user_embed_hist; q = item_embed_global;
        W = proj_i_w; bias = proj_i_b; etarget = item_embed_target;
    }

    // ---- stage 1a: history indices + mask ----------------------------------
    if (tid < LPAD) {
        int idx; float m;
        if (tid < LHIST) {
            idx = hist[(size_t)target * LHIST + tid];
            m   = (idx == counterpart || idx == pad_val) ? 1.0f : 0.0f;
        } else {                       // padding rows: point at zeroed pad row
            idx = pad_val; m = 1.0f;
        }
        sIdx[tid] = idx;
        sMask[tid] = m;
    }
    __syncthreads();

    // ---- stage 1b: gather refer rows -> LDS (f32 -> f16) -------------------
    // one wave per row; each lane moves one float4 (4 elems of 128)
    for (int l = wave; l < LPAD; l += 8) {
        const int idx = sIdx[l];
        const float4 v = ((const float4*)(ehist + (size_t)idx * DMODEL))[lane];
        h2 lo; lo[0] = (_Float16)v.x; lo[1] = (_Float16)v.y;
        h2 hi; hi[0] = (_Float16)v.z; hi[1] = (_Float16)v.w;
        _Float16* p = &sRef[l * ROWW + lane * 4];
        *(h2*)p       = lo;
        *(h2*)(p + 2) = hi;
    }
    __syncthreads();

    // ---- stage 2: K = tanh(refer @ W^T + b); score = q . K via WMMA --------
    const int mrow  = lane & 15;     // row within tile (A) / column n (B,C)
    const int khalf = lane >> 4;     // K-half selector per 16-bit layout table

    for (int rt = wave; rt < NT_ROW; rt += 8) {
        const int l0 = rt * 16;

        // A fragments for this row tile, all 4 k-steps (16x32 f16 each)
        v16h afrag[4];
        #pragma unroll
        for (int kt = 0; kt < 4; ++kt) {
            #pragma unroll
            for (int j = 0; j < 8; ++j) {
                const int k = kt * 32 + ((j >= 4) ? 16 : 0) + khalf * 8 + 2 * (j & 3);
                const h2 p = *(const h2*)&sRef[(l0 + mrow) * ROWW + k];
                afrag[kt][2 * j]     = p[0];
                afrag[kt][2 * j + 1] = p[1];
            }
        }

        float rowScore[8];
        #pragma unroll
        for (int r = 0; r < 8; ++r) rowScore[r] = 0.0f;

        #pragma unroll
        for (int nt = 0; nt < 8; ++nt) {
            const int n = nt * 16 + mrow;
            v8f acc = {};
            #pragma unroll
            for (int kt = 0; kt < 4; ++kt) {
                v16h bfrag;        // B[k][n] = W[n][k]  (K = refer @ W^T)
                #pragma unroll
                for (int j = 0; j < 8; ++j) {
                    const int k = kt * 32 + ((j >= 4) ? 16 : 0) + khalf * 8 + 2 * (j & 3);
                    const float2 w2 = *(const float2*)&W[(size_t)n * DMODEL + k];
                    bfrag[2 * j]     = (_Float16)w2.x;
                    bfrag[2 * j + 1] = (_Float16)w2.y;
                }
                acc = __builtin_amdgcn_wmma_f32_16x16x32_f16(
                        false, afrag[kt], false, bfrag, (short)0, acc, false, false);
            }
            const float bn = bias[n];
            const float qn = q[n];
            #pragma unroll
            for (int r = 0; r < 8; ++r)
                rowScore[r] += qn * fast_tanh(acc[r] + bn);   // bias+tanh+q·K
        }

        // per-row score is spread over the 16 lanes of each half: reduce
        #pragma unroll
        for (int m = 1; m <= 8; m <<= 1) {
            #pragma unroll
            for (int r = 0; r < 8; ++r)
                rowScore[r] += __shfl_xor(rowScore[r], m, 32);
        }
        if ((lane & 15) == 0) {
            const int base = l0 + khalf * 8;   // lanes16-31 hold rows M=8..15
            #pragma unroll
            for (int r = 0; r < 8; ++r) {
                const int l = base + r;
                sScore[l] = (sMask[l] > 0.0f) ? -1e9f : rowScore[r] * SCALE;
            }
        }
    }
    __syncthreads();

    // ---- stage 3: softmax normalization (wave 0) ---------------------------
    if (wave == 0) {
        float mx = -INFINITY;
        for (int l = lane; l < LPAD; l += 32) mx = fmaxf(mx, sScore[l]);
        #pragma unroll
        for (int m = 1; m < 32; m <<= 1) mx = fmaxf(mx, __shfl_xor(mx, m, 32));
        float sum = 0.0f;
        for (int l = lane; l < LPAD; l += 32) {
            const float e = __expf(sScore[l] - mx);
            sScore[l] = e;
            sum += e;
        }
        #pragma unroll
        for (int m = 1; m < 32; m <<= 1) sum += __shfl_xor(sum, m, 32);
        if (lane == 0) sInv[0] = 1.0f / sum;
    }
    __syncthreads();

    // ---- stage 4: out_d = sum_l attn_l * refer[l][d], + id-embed scatter ---
    const float inv = sInv[0];
    {
        const int d    = tid & (DMODEL - 1);
        const int half = tid >> 7;             // split L range across 2 halves
        float acc = 0.0f;
        const int lb = half * (LPAD / 2);
        for (int l = lb; l < lb + LPAD / 2; ++l)
            acc += sScore[l] * (float)sRef[l * ROWW + d];
        sPartial[half * DMODEL + d] = acc;
    }
    __syncthreads();

    if (tid < DMODEL) {
        const float hval = (sPartial[tid] + sPartial[DMODEL + tid]) * inv;
        const long long r256 = (long long)b * 256;
        float* id_cat   = out;
        float* hist_cat = out + (long long)Bn * 256;
        float* user_cat = out + 2ll * Bn * 256;
        float* item_cat = out + 3ll * Bn * 256;
        float* side_cat = side ? item_cat : user_cat;

        hist_cat[r256 + side * DMODEL + tid] = hval;          // hist_cat half
        side_cat[r256 + DMODEL + tid]        = hval;          // {user,item}_cat hist half

        const float idv = etarget[(size_t)target * DMODEL + tid];
        id_cat[r256 + side * DMODEL + tid]   = idv;           // id_cat half
        side_cat[r256 + tid]                 = idv;           // {user,item}_cat id half
    }
}

extern "C" void kernel_launch(void* const* d_in, const int* in_sizes, int n_in,
                              void* d_out, int out_size, void* d_ws, size_t ws_size,
                              hipStream_t stream) {
    const int*   user_idx          = (const int*)d_in[0];
    const int*   item_idx          = (const int*)d_in[1];
    const int*   user_hist         = (const int*)d_in[2];
    const int*   item_hist         = (const int*)d_in[3];
    const float* user_embed_target = (const float*)d_in[4];
    const float* user_embed_hist   = (const float*)d_in[5];
    const float* user_embed_global = (const float*)d_in[6];
    const float* item_embed_target = (const float*)d_in[7];
    const float* item_embed_hist   = (const float*)d_in[8];
    const float* item_embed_global = (const float*)d_in[9];
    const float* proj_u_w          = (const float*)d_in[10];
    const float* proj_u_b          = (const float*)d_in[11];
    const float* proj_i_w          = (const float*)d_in[12];
    const float* proj_i_b          = (const float*)d_in[13];

    const int Bn      = in_sizes[0];
    const int n_users = in_sizes[2] / LHIST;
    const int n_items = in_sizes[3] / LHIST;

    dim3 grid(Bn, 2), block(256);
    din_attn_kernel<<<grid, block, 0, stream>>>(
        user_idx, item_idx, user_hist, item_hist,
        user_embed_target, user_embed_hist, user_embed_global,
        item_embed_target, item_embed_hist, item_embed_global,
        proj_u_w, proj_u_b, proj_i_w, proj_i_b,
        (float*)d_out, Bn, n_users, n_items);
}
